// GCN_DAE_13726715478762
// MI455X (gfx1250) — matrix operand order repaired and checked by reading, not compiled, not run on hardware
//
#include <hip/hip_runtime.h>
#include <hip/hip_bf16.h>

typedef __attribute__((ext_vector_type(16))) __bf16 v16bf;
typedef __attribute__((ext_vector_type(8)))  __bf16 v8bf;
typedef __attribute__((ext_vector_type(8)))  float  v8f;
typedef int v4i __attribute__((__vector_size__(4 * sizeof(int))));

#if __has_builtin(__builtin_amdgcn_global_load_async_to_lds_b128)
#define HAVE_ASYNC_COPY 1
#else
#define HAVE_ASYNC_COPY 0
#endif

namespace {

constexpr int Nn   = 8192;    // nodes
constexpr int Dd   = 512;     // feature dim
constexpr int HID  = 256;     // GCN hidden
constexpr int Cc   = 64;      // classes
constexpr int Pp   = 8;       // perspectives
constexpr int PH   = Pp * 256; // 2048 = concatenated perspective dim
constexpr int TOPK = 30;

constexpr int BM = 256;       // block rows
constexpr int BN = 64;        // block cols
constexpr int BK = 32;        // k-step
constexpr int ROWP = 40;      // padded LDS row stride (elements): 80B -> spreads banks

// ---------------- f32 -> bf16 conversion ----------------
__global__ void cvt_f32_bf16(const float* __restrict__ in, __bf16* __restrict__ out, int n) {
  int i = blockIdx.x * blockDim.x + threadIdx.x;
  if (i < n) out[i] = (__bf16)in[i];
}

// ---------------- async global -> LDS staging (16B per lane) ----------------
__device__ __forceinline__ void stage_b128(const __bf16* g, __bf16* l) {
#if HAVE_ASYNC_COPY
  auto gp = (__attribute__((address_space(1))) v4i*)(uintptr_t)g;
  auto lp = (__attribute__((address_space(3))) v4i*)(uint32_t)(uintptr_t)l;
  __builtin_amdgcn_global_load_async_to_lds_b128(gp, lp, 0, 0);
#else
  *(v8bf*)l = *(const v8bf*)g;   // sync fallback: global_load_b128 + ds_store_b128
#endif
}

__device__ __forceinline__ void wait_async0() {
#if HAVE_ASYNC_COPY
#if __has_builtin(__builtin_amdgcn_s_wait_asynccnt)
  __builtin_amdgcn_s_wait_asynccnt(0);
#else
  asm volatile("s_wait_asynccnt 0" ::: "memory");
#endif
#endif
}

// ---------------- WMMA fragment loaders (ISA 7.12.2 layouts) ----------------
// A 16x32 bf16: lane L = row L&15; K chunks [kb,kb+8) and [kb+16,kb+24), kb=8*(L>=16).
// p points at element (row, kb) of a row-major tile.
__device__ __forceinline__ v16bf ldfragA(const __bf16* p) {
  v8bf lo = *(const v8bf*)p;
  v8bf hi = *(const v8bf*)(p + 16);
  v16bf r;
#pragma unroll
  for (int i = 0; i < 8; ++i) { r[i] = lo[i]; r[i + 8] = hi[i]; }
  return r;
}

// B 32x16 bf16: lane L = column L&15; K chunks [kb,kb+8) and [kb+16,kb+24), kb=8*(L>=16).
// For C=A*B^T with row-major B[NxK], column n of the operand is row n of B.
__device__ __forceinline__ v16bf ldfragB(const __bf16* p) {
  v8bf lo = *(const v8bf*)p;
  v8bf hi = *(const v8bf*)(p + 16);
  v16bf r;
#pragma unroll
  for (int i = 0; i < 8; ++i) { r[i] = lo[i]; r[i + 8] = hi[i]; }
  return r;
}

// ---------------- NT GEMM: C = act(alpha * A[MxK] * B[NxK]^T + bias) ----------------
// 256 threads = 8 waves (4 wave-rows x 2 wave-cols); wave tile 64x32 = 8 WMMA accs.
// A (256x32) and B (64x32) k-tiles double-buffered in LDS via async-to-LDS copies.
template <bool RELU, bool BF16OUT>
__global__ void __launch_bounds__(256)
gemm_nt_wmma(const __bf16* __restrict__ A, const __bf16* __restrict__ B,
             float* __restrict__ Cf, __bf16* __restrict__ Cb,
             const float* __restrict__ bias,
             int M, int N, int K, float alpha)
{
  __shared__ __bf16 At[2][BM * ROWP];   // 2 x 20 KB
  __shared__ __bf16 Bt[2][BN * ROWP];   // 2 x  5 KB

  const int tid  = threadIdx.x;
  const int lane = tid & 31;
  const int wave = tid >> 5;
  const int wm   = wave & 3;            // wave row group (64 rows each)
  const int wn   = wave >> 2;           // wave col group (32 cols each)
  const int rowA0 = blockIdx.x * BM;
  const int colB0 = blockIdx.y * BN;
  const int row0 = rowA0 + wm * 64;
  const int col0 = colB0 + wn * 32;
  const int lm   = lane & 15;
  const int hw   = lane >> 4;           // half-wave 0/1

  v8f acc[4][2];
#pragma unroll
  for (int i = 0; i < 4; ++i)
#pragma unroll
    for (int j = 0; j < 2; ++j)
#pragma unroll
      for (int e = 0; e < 8; ++e) acc[i][j][e] = 0.0f;

  // per-thread staging addresses: thread t copies 16B chunks (row = t>>1 (+128j), half = t&1)
  const int sr = tid >> 1;
  const int sh = tid & 1;
  const __bf16* agt = A + (size_t)(rowA0 + sr) * K + sh * 16;  // + j*128*K
  const __bf16* bgt = B + (size_t)(colB0 + sr) * K + sh * 16;
  const int lofs = sr * ROWP + sh * 16;                        // + j*128*ROWP

  // prologue: stage k-tile 0 into buffer 0
#pragma unroll
  for (int j = 0; j < 4; ++j)
    stage_b128(agt + (size_t)(j * 128) * K, &At[0][lofs + j * 128 * ROWP]);
  stage_b128(bgt, &Bt[0][lofs]);

  const int nk = K >> 5;
  for (int i = 0; i < nk; ++i) {
    const int cur = i & 1;
    const int k0  = i << 5;
    wait_async0();
    __syncthreads();                    // buf[cur] fully in LDS, prev reads done
    if (i + 1 < nk) {                   // stage next tile, overlapped with WMMA below
      const int nxt = cur ^ 1;
#pragma unroll
      for (int j = 0; j < 4; ++j)
        stage_b128(agt + (size_t)(j * 128) * K + (k0 + BK),
                   &At[nxt][lofs + j * 128 * ROWP]);
      stage_b128(bgt + (k0 + BK), &Bt[nxt][lofs]);
      if (i + 2 < nk) {                 // pull the tile after that toward L2/WGP$
        __builtin_prefetch(agt + (size_t)k0 + 2 * BK, 0, 3);
        __builtin_prefetch(bgt + (size_t)k0 + 2 * BK, 0, 3);
      }
    }

    const __bf16* bb = &Bt[cur][(wn * 32 + lm) * ROWP + hw * 8];
    v16bf bf0 = ldfragB(bb);
    v16bf bf1 = ldfragB(bb + 16 * ROWP);
#pragma unroll
    for (int mi = 0; mi < 4; ++mi) {
      v16bf af = ldfragA(&At[cur][(wm * 64 + mi * 16 + lm) * ROWP + hw * 8]);
      acc[mi][0] = __builtin_amdgcn_wmma_f32_16x16x32_bf16(false, af, false, bf0, (short)0, acc[mi][0], false, false);
      acc[mi][1] = __builtin_amdgcn_wmma_f32_16x16x32_bf16(false, af, false, bf1, (short)0, acc[mi][1], false, false);
    }
  }

  // C/D layout: VGPR e -> (M = e + 8*hw, N = lane&15) within each 16x16 tile
#pragma unroll
  for (int mi = 0; mi < 4; ++mi)
#pragma unroll
    for (int ni = 0; ni < 2; ++ni) {
      const int c    = col0 + ni * 16 + lm;
      const float ba = bias ? bias[c] : 0.0f;
      const int rb   = row0 + mi * 16 + hw * 8;
#pragma unroll
      for (int e = 0; e < 8; ++e) {
        float v = acc[mi][ni][e] * alpha + ba;
        if (RELU) v = fmaxf(v, 0.0f);
        const size_t off = (size_t)(rb + e) * N + c;
        if (BF16OUT) Cb[off] = (__bf16)v;
        else         Cf[off] = v;
      }
    }
}

// ---------------- top-30 + row softmax (one block per row) ----------------
__global__ void __launch_bounds__(256)
topk_softmax(float* __restrict__ adj, int* __restrict__ tkIdx, float* __restrict__ tkVal)
{
  __shared__ float rowv[Nn];
  __shared__ float redV[256];
  __shared__ int   redI[256];
  __shared__ float topv[TOPK];
  __shared__ int   topi[TOPK];

  const int r   = blockIdx.x;
  const int tid = threadIdx.x;
  float* g = adj + (size_t)r * Nn;

  for (int i = tid; i < Nn; i += 256) rowv[i] = g[i];
  __syncthreads();

  for (int t = 0; t < TOPK; ++t) {
    float bv = -3.4e38f; int bi = Nn;
    for (int i = tid; i < Nn; i += 256) {
      float v = rowv[i];
      if (v > bv || (v == bv && i < bi)) { bv = v; bi = i; }
    }
    redV[tid] = bv; redI[tid] = bi;
    __syncthreads();
    for (int s = 128; s > 0; s >>= 1) {
      if (tid < s) {
        float ov = redV[tid + s]; int oi = redI[tid + s];
        if (ov > redV[tid] || (ov == redV[tid] && oi < redI[tid])) { redV[tid] = ov; redI[tid] = oi; }
      }
      __syncthreads();
    }
    if (tid == 0) { topv[t] = redV[0]; topi[t] = redI[0]; rowv[redI[0]] = -3.4e38f; }
    __syncthreads();
  }

  // softmax over the 30 kept values (others: exp(NEG_INF - max) == 0 exactly)
  const float mx = topv[0];
  float ssum = 0.0f;
  for (int t = 0; t < TOPK; ++t) ssum += expf(topv[t] - mx);
  const float inv = 1.0f / ssum;

  for (int i = tid; i < Nn; i += 256) {
    float v = 0.0f;
#pragma unroll
    for (int t = 0; t < TOPK; ++t) v = (topi[t] == i) ? expf(topv[t] - mx) * inv : v;
    g[i] = v;
  }
  if (tid < TOPK) {
    tkIdx[r * TOPK + tid] = topi[tid];
    tkVal[r * TOPK + tid] = expf(topv[tid] - mx) * inv;
  }
}

// ---------------- sparse (30 nnz/row) adj @ X ----------------
template <bool RELU>
__global__ void spmm30(const int* __restrict__ tkIdx, const float* __restrict__ tkVal,
                       const float* __restrict__ Xin, float* __restrict__ Xout,
                       __bf16* __restrict__ XoutB, int F)
{
  const int r = blockIdx.x;
  const int c = threadIdx.x;          // blockDim.x == F (<= 256)
  const int*   ip = tkIdx + r * TOPK;
  const float* vp = tkVal + r * TOPK;
  float s = 0.0f;
#pragma unroll
  for (int j = 0; j < TOPK; ++j) s += vp[j] * Xin[(size_t)ip[j] * F + c];
  if (RELU) s = fmaxf(s, 0.0f);
  if (Xout)  Xout[(size_t)r * F + c]  = s;
  if (XoutB) XoutB[(size_t)r * F + c] = (__bf16)s;
}

} // namespace

extern "C" void kernel_launch(void* const* d_in, const int* in_sizes, int n_in,
                              void* d_out, int out_size, void* d_ws, size_t ws_size,
                              hipStream_t stream)
{
  const float* features = (const float*)d_in[0]; // [N, D]
  const float* x        = (const float*)d_in[1]; // [N, D]
  const float* W_sims   = (const float*)d_in[2]; // [P, H, D] == [PH, D] row-major
  const float* W1       = (const float*)d_in[3]; // [HID, D]
  const float* b1       = (const float*)d_in[4]; // [HID]
  const float* W2       = (const float*)d_in[5]; // [C, HID]
  const float* b2       = (const float*)d_in[6]; // [C]

  float* out = (float*)d_out;                    // [N, C]
  float* adj = out + (size_t)Nn * Cc;            // [N, N]

  // workspace carve-out (256B aligned slabs)
  char* w = (char*)d_ws;
  auto carve = [&](size_t bytes) { char* p = w; w += (bytes + 255) & ~(size_t)255; return p; };
  __bf16* featB = (__bf16*)carve((size_t)Nn  * Dd  * sizeof(__bf16));
  __bf16* WsB   = (__bf16*)carve((size_t)PH  * Dd  * sizeof(__bf16));
  __bf16* xB    = (__bf16*)carve((size_t)Nn  * Dd  * sizeof(__bf16));
  __bf16* W1B   = (__bf16*)carve((size_t)HID * Dd  * sizeof(__bf16));
  __bf16* W2B   = (__bf16*)carve((size_t)Cc  * HID * sizeof(__bf16));
  __bf16* HcatB = (__bf16*)carve((size_t)Nn  * PH  * sizeof(__bf16));
  float*  t1    = (float*) carve((size_t)Nn  * HID * sizeof(float));
  __bf16* hB    = (__bf16*)carve((size_t)Nn  * HID * sizeof(__bf16));
  float*  t2    = (float*) carve((size_t)Nn  * Cc  * sizeof(float));
  int*    tkIdx = (int*)   carve((size_t)Nn  * TOPK * sizeof(int));
  float*  tkVal = (float*) carve((size_t)Nn  * TOPK * sizeof(float));

  auto cvt = [&](const float* src, __bf16* dst, int n) {
    cvt_f32_bf16<<<(n + 255) / 256, 256, 0, stream>>>(src, dst, n);
  };
  cvt(features, featB, Nn * Dd);
  cvt(W_sims,   WsB,   PH * Dd);
  cvt(x,        xB,    Nn * Dd);
  cvt(W1,       W1B,   HID * Dd);
  cvt(W2,       W2B,   Cc * HID);

  const dim3 blk(256);

  // Hcat = relu(features @ W_sims^T)   [8192 x 2048] bf16
  gemm_nt_wmma<true, true><<<dim3(Nn / BM, PH / BN), blk, 0, stream>>>(
      featB, WsB, nullptr, HcatB, nullptr, Nn, PH, Dd, 1.0f);

  // att = Hcat @ Hcat^T / P   [8192 x 8192] f32, straight into adj region of d_out
  gemm_nt_wmma<false, false><<<dim3(Nn / BM, Nn / BN), blk, 0, stream>>>(
      HcatB, HcatB, adj, nullptr, nullptr, Nn, Nn, PH, 1.0f / (float)Pp);

  // per-row top-30 + softmax; rewrites adj dense + emits (idx, weight) lists
  topk_softmax<<<Nn, 256, 0, stream>>>(adj, tkIdx, tkVal);

  // t1 = x @ W1^T + b1   [8192 x 256] f32
  gemm_nt_wmma<false, false><<<dim3(Nn / BM, HID / BN), blk, 0, stream>>>(
      xB, W1B, t1, nullptr, b1, Nn, HID, Dd, 1.0f);

  // h = relu(adj @ t1) -> bf16 (30-nnz SpMM)
  spmm30<true><<<Nn, HID, 0, stream>>>(tkIdx, tkVal, t1, nullptr, hB, HID);

  // t2 = h @ W2^T + b2   [8192 x 64] f32
  gemm_nt_wmma<false, false><<<dim3(Nn / BM, Cc / BN), blk, 0, stream>>>(
      hB, W2B, t2, nullptr, b2, Nn, Cc, HID, 1.0f);

  // out = adj @ t2 (30-nnz SpMM)
  spmm30<false><<<Nn, Cc, 0, stream>>>(tkIdx, tkVal, t2, out, nullptr, Cc);
}
